// SelfAttention_60241211293951
// MI455X (gfx1250) — matrix operand order repaired and checked by reading, not compiled, not run on hardware
//
#include <hip/hip_runtime.h>

// ---------------------------------------------------------------------------
// Self-attention forward for MI455X (gfx1250, wave32, WMMA bf16, TDM)
// B=2, T=2048, C=1024, H=16, Dh=64
// ---------------------------------------------------------------------------

#define B_SZ   2
#define T_SEQ  2048
#define C_DIM  1024
#define H_NUM  16
#define D_HEAD 64
#define M_TOT  (B_SZ * T_SEQ)   // 4096

typedef __bf16 v16bf __attribute__((ext_vector_type(16)));
typedef float  v8f   __attribute__((ext_vector_type(8)));
typedef unsigned int u32x4 __attribute__((ext_vector_type(4)));
typedef int          i32x4 __attribute__((ext_vector_type(4)));
typedef int          i32x8 __attribute__((ext_vector_type(8)));

#if defined(__has_builtin)
#if __has_builtin(__builtin_amdgcn_tensor_load_to_lds) && __has_builtin(__builtin_amdgcn_s_wait_tensorcnt)
#define USE_TDM 1
#endif
#endif
#ifndef USE_TDM
#define USE_TDM 0
#endif

union Frag16 {
    v16bf v;
    uint4 q[2];
    unsigned short u[16];
};

__device__ __forceinline__ unsigned short f2bf(float f) {
    unsigned u = __builtin_bit_cast(unsigned, f);
    u += 0x7FFFu + ((u >> 16) & 1u);          // round-to-nearest-even
    return (unsigned short)(u >> 16);
}

#if USE_TDM
// Issue one TDM 2D tile load: 128 rows x 32 bf16 elements, row stride = K elems.
// D# layout per CDNA5 ISA.
__device__ __forceinline__ void tdm_load_tile(unsigned lds_byte_addr,
                                              const void* gptr, int K_elems) {
    unsigned long long ga = (unsigned long long)gptr;
    u32x4 g0;
    g0.x = 1u;                                         // count=1 (valid), 2D, no gather
    g0.y = lds_byte_addr;                              // lds_addr (bytes)
    g0.z = (unsigned)(ga & 0xFFFFFFFFu);               // global_addr[31:0]
    g0.w = (unsigned)((ga >> 32) & 0x1FFFFFFu)         // global_addr[56:32]
           | 0x80000000u;                              // type=2 ("image")
    i32x8 g1;
    g1[0] = (1 << 16);                                 // wg_mask=0, data_size=1 (2B)
    g1[1] = (int)((K_elems & 0xFFFF) << 16);           // tensor_dim0[15:0] @ bits63:48
    g1[2] = (int)(((unsigned)K_elems >> 16) & 0xFFFFu) // tensor_dim0[31:16]
          | (128 << 16);                               // tensor_dim1[15:0] = 128 rows
    g1[3] = (32 << 16);                                // tensor_dim1 hi=0 | tile_dim0=32
    g1[4] = 128;                                       // tile_dim1=128, tile_dim2=0
    g1[5] = K_elems;                                   // tensor_dim0_stride[31:0]
    g1[6] = 0;                                         // stride hi / dim1_stride lo
    g1[7] = 0;
    i32x4 z4 = {0, 0, 0, 0};
#if defined(__clang_major__) && (__clang_major__ >= 23)
    i32x8 z8 = {0, 0, 0, 0, 0, 0, 0, 0};
    __builtin_amdgcn_tensor_load_to_lds(g0, g1, z4, z4, z8, 0);
#else
    __builtin_amdgcn_tensor_load_to_lds(g0, g1, z4, z4, 0);
#endif
}
#endif

// ---------------------------------------------------------------------------
// fp32 -> bf16(bits) conversion
// ---------------------------------------------------------------------------
__global__ void f32_to_bf16_kernel(const float* __restrict__ in,
                                   unsigned short* __restrict__ out, int n) {
    int i = blockIdx.x * blockDim.x + threadIdx.x;
    int stride = gridDim.x * blockDim.x;
    for (; i < n; i += stride) out[i] = f2bf(in[i]);
}

// ---------------------------------------------------------------------------
// bf16 GEMM:  Y[m,n] = sum_k A[m,k] * W[n,k]   (i.e. x @ W.T)
// Block tile 128x128, 8 waves (2x4), wave tile 64x32, K-step 32.
// Double-buffered LDS tiles filled by the Tensor Data Mover, TENSORcnt sync.
// mode 0: Cf[m,n] = y (f32)   mode 1: RoPE -> Cb=[B,H,T,D]   mode 2: Cb=[B,H,D,T]
// ---------------------------------------------------------------------------
__global__ __launch_bounds__(256) void gemm_bf16_kernel(
    const unsigned short* __restrict__ A,
    const unsigned short* __restrict__ W,
    int M, int N, int K, int mode,
    float* __restrict__ Cf, unsigned short* __restrict__ Cb,
    const float* __restrict__ cosT, const float* __restrict__ sinT) {

    __shared__ union {
        struct {
            unsigned short a[2][128][32];   // 16 KB (double-buffered)
            unsigned short b[2][128][32];   // 16 KB
        } s;
        float c[128][128];                  // 64 KB epilogue staging
    } sm;

    const int tid    = threadIdx.x;
    const int lane   = tid & 31;
    const int ln     = lane & 15;
    const int hh     = lane >> 4;
    const int waveId = tid >> 5;
    const int wm     = (waveId >> 2) * 64;   // wave M offset (0 / 64)
    const int wn     = (waveId & 3) * 32;    // wave N offset (0..96)
    const int blockM = blockIdx.y * 128;
    const int blockN = blockIdx.x * 128;

    const v8f zero8 = {0.f, 0.f, 0.f, 0.f, 0.f, 0.f, 0.f, 0.f};
    v8f acc[4][2];
#pragma unroll
    for (int mt = 0; mt < 4; ++mt)
#pragma unroll
        for (int nt = 0; nt < 2; ++nt) acc[mt][nt] = zero8;

#if USE_TDM
    // prime buffer 0 via TDM (one issuing thread; TDM ignores EXEC)
    if (tid == 0) {
        tdm_load_tile((unsigned)(uintptr_t)&sm.s.a[0][0][0],
                      &A[(size_t)blockM * K], K);
        tdm_load_tile((unsigned)(uintptr_t)&sm.s.b[0][0][0],
                      &W[(size_t)blockN * K], K);
    }
#endif

    for (int kk = 0, it = 0; kk < K; kk += 32, ++it) {
        const int buf = it & 1;
        const bool more = (kk + 32) < K;

#if USE_TDM
        // kick DMA for the next K-slice into the other buffer, then wait for
        // the current pair (in-order TENSORcnt: <=2 leaves only the new pair).
        if (tid == 0 && more) {
            tdm_load_tile((unsigned)(uintptr_t)&sm.s.a[buf ^ 1][0][0],
                          &A[(size_t)blockM * K + kk + 32], K);
            tdm_load_tile((unsigned)(uintptr_t)&sm.s.b[buf ^ 1][0][0],
                          &W[(size_t)blockN * K + kk + 32], K);
        }
        if (more) __builtin_amdgcn_s_wait_tensorcnt(2);
        else      __builtin_amdgcn_s_wait_tensorcnt(0);
        __syncthreads();
#else
        // fallback: cooperative vector-load staging
#pragma unroll
        for (int i = tid; i < 512; i += 256) {
            int row = i >> 2;
            int col = (i & 3) << 3;
            *(uint4*)&sm.s.a[buf][row][col] =
                *(const uint4*)&A[(size_t)(blockM + row) * K + kk + col];
            *(uint4*)&sm.s.b[buf][row][col] =
                *(const uint4*)&W[(size_t)(blockN + row) * K + kk + col];
        }
        __syncthreads();
#endif

        // A fragments: lane ln = M row, hh selects K chunks {0..7,16..23}/{8..15,24..31}
        Frag16 af[4], bf2[2];
#pragma unroll
        for (int mt = 0; mt < 4; ++mt) {
            const unsigned short* p = &sm.s.a[buf][wm + mt * 16 + ln][hh * 8];
            af[mt].q[0] = *(const uint4*)p;
            af[mt].q[1] = *(const uint4*)(p + 16);
        }
        // B fragments: lane ln = N row, hh selects K chunks {0..7,16..23}/{8..15,24..31}
#pragma unroll
        for (int nt = 0; nt < 2; ++nt) {
            const unsigned short* p = &sm.s.b[buf][wn + nt * 16 + ln][hh * 8];
            bf2[nt].q[0] = *(const uint4*)p;
            bf2[nt].q[1] = *(const uint4*)(p + 16);
        }
#pragma unroll
        for (int mt = 0; mt < 4; ++mt)
#pragma unroll
            for (int nt = 0; nt < 2; ++nt)
                acc[mt][nt] = __builtin_amdgcn_wmma_f32_16x16x32_bf16(
                    false, af[mt].v, false, bf2[nt].v, (short)0, acc[mt][nt],
                    false, false);
        __syncthreads();   // all waves done reading buf before TDM refills it
    }

    if (mode == 0) {
        // plain f32 store: D-layout element (r,hh) -> row mt*16 + hh*8 + r
#pragma unroll
        for (int mt = 0; mt < 4; ++mt)
#pragma unroll
            for (int nt = 0; nt < 2; ++nt)
#pragma unroll
                for (int r = 0; r < 8; ++r) {
                    int m = blockM + wm + mt * 16 + hh * 8 + r;
                    int n = blockN + wn + nt * 16 + ln;
                    Cf[(size_t)m * N + n] = acc[mt][nt][r];
                }
        return;
    }

    // stage f32 tile in LDS for RoPE pairing / transpose store
    __syncthreads();
#pragma unroll
    for (int mt = 0; mt < 4; ++mt)
#pragma unroll
        for (int nt = 0; nt < 2; ++nt)
#pragma unroll
            for (int r = 0; r < 8; ++r)
                sm.c[wm + mt * 16 + hh * 8 + r][wn + nt * 16 + ln] = acc[mt][nt][r];
    __syncthreads();

    for (int i = tid; i < 128 * 128; i += 256) {
        int ml = i >> 7, nl = i & 127;
        int gm = blockM + ml, gn = blockN + nl;
        int t = gm & (T_SEQ - 1);
        int bb = gm >> 11;                 // gm / T_SEQ
        int hd = gn >> 6, d = gn & 63;
        float y = sm.c[ml][nl];
        if (mode == 1) {
            float y2 = sm.c[ml][nl ^ 32];  // partner d +/- 32 (same head, in-tile)
            float cc = cosT[t * 64 + d];
            float ss = sinT[t * 64 + d];
            float rot = (d < 32) ? -y2 : y2;
            y = y * cc + rot * ss;
            Cb[(((size_t)bb * H_NUM + hd) * T_SEQ + t) * 64 + d] = f2bf(y);
        } else { // mode 2: V transposed -> [B,H,D,T]
            Cb[(((size_t)bb * H_NUM + hd) * 64 + d) * T_SEQ + t] = f2bf(y);
        }
    }
}

// ---------------------------------------------------------------------------
// Flash attention (causal, online softmax), bf16 WMMA.
// Grid: (T/64, H, B), block 128 = 4 waves, each wave owns 16 q-rows.
// S' = K_tile @ Q^T so P lands directly in the A-fragment layout for P@V.
// ---------------------------------------------------------------------------
__global__ __launch_bounds__(128) void flash_attn_kernel(
    const unsigned short* __restrict__ Q,   // [B,H,T,D] bf16 (rope'd)
    const unsigned short* __restrict__ Kt,  // [B,H,T,D] bf16 (rope'd)
    const unsigned short* __restrict__ VT,  // [B,H,D,T] bf16
    unsigned short* __restrict__ O,         // [B,T,H,D] bf16
    float scale) {

    const int qblk = blockIdx.x;
    const int head = blockIdx.y;
    const int b    = blockIdx.z;
    const int wave = threadIdx.x >> 5;
    const int lane = threadIdx.x & 31;
    const int ln   = lane & 15;
    const int hh   = lane >> 4;
    const int qbase = qblk * 64 + wave * 16;

    const unsigned short* Qh = Q  + ((size_t)(b * H_NUM + head) * T_SEQ) * 64;
    const unsigned short* Kh = Kt + ((size_t)(b * H_NUM + head) * T_SEQ) * 64;
    const unsigned short* Vh = VT + ((size_t)(b * H_NUM + head) * 64) * T_SEQ;

    // Q as B-operand (dh x 16q): lane = q column, dh chunks {0..7,16..23}/{8..15,24..31}
    Frag16 qf[2];
#pragma unroll
    for (int f = 0; f < 2; ++f) {
        const unsigned short* p = &Qh[(size_t)(qbase + ln) * 64 + f * 32 + hh * 8];
        qf[f].q[0] = *(const uint4*)p;
        qf[f].q[1] = *(const uint4*)(p + 16);
    }

    const v8f zero8 = {0.f, 0.f, 0.f, 0.f, 0.f, 0.f, 0.f, 0.f};
    v8f o[4];
#pragma unroll
    for (int dt = 0; dt < 4; ++dt) o[dt] = zero8;
    float mrow = -1e30f, lrow = 0.f;

    const int nsteps = qblk * 2 + 2;   // kv covered: [0, qblock_end)
    for (int j = 0; j < nsteps; ++j) {
        const int kv0 = j * 32;

        // prefetch next kv step's K rows and V columns (global_prefetch_b8)
        if (j + 1 < nsteps) {
            __builtin_prefetch(&Kh[(size_t)(kv0 + 32 + ln) * 64], 0, 3);
            __builtin_prefetch(&Vh[(size_t)ln * T_SEQ + kv0 + 32], 0, 3);
        }

        // S'[kv,q] = K @ Q^T : two 16-kv subtiles, each with 2 dh-step WMMAs
        v8f sAcc[2];
#pragma unroll
        for (int st = 0; st < 2; ++st) {
            sAcc[st] = zero8;
#pragma unroll
            for (int f = 0; f < 2; ++f) {
                Frag16 kf;  // K rows as A-operand: lane ln = kv row
                const unsigned short* p =
                    &Kh[(size_t)(kv0 + st * 16 + ln) * 64 + f * 32 + hh * 8];
                kf.q[0] = *(const uint4*)p;
                kf.q[1] = *(const uint4*)(p + 16);
                sAcc[st] = __builtin_amdgcn_wmma_f32_16x16x32_bf16(
                    false, kf.v, false, qf[f].v, (short)0, sAcc[st], false, false);
            }
        }

        // scale + causal mask + online softmax (stats per q = per lane ln)
        const int qg = qbase + ln;
        float sv[2][8];
        float smax = -1e30f;
#pragma unroll
        for (int st = 0; st < 2; ++st)
#pragma unroll
            for (int r = 0; r < 8; ++r) {
                float x = sAcc[st][r] * scale;
                int kv = kv0 + st * 16 + hh * 8 + r;
                if (kv > qg) x = -1e30f;
                sv[st][r] = x;
                smax = fmaxf(smax, x);
            }
        smax = fmaxf(smax, __shfl_xor(smax, 16, 32));
        float mnew = fmaxf(mrow, smax);
        float sc = __expf(mrow - mnew);
        mrow = mnew;

        Frag16 pf;
        float lsum = 0.f;
#pragma unroll
        for (int st = 0; st < 2; ++st)
#pragma unroll
            for (int r = 0; r < 8; ++r) {
                float p = __expf(sv[st][r] - mnew);
                lsum += p;
                pf.u[st * 8 + r] = f2bf(p);   // exactly A-fragment element order
            }
        lsum += __shfl_xor(lsum, 16, 32);
        lrow = lrow * sc + lsum;

        // rescale O: row q' = hh*8 + r; stats live in lane q' (dup per half)
        float scr[8];
#pragma unroll
        for (int r = 0; r < 8; ++r) scr[r] = __shfl(sc, hh * 8 + r, 32);
#pragma unroll
        for (int dt = 0; dt < 4; ++dt)
#pragma unroll
            for (int r = 0; r < 8; ++r) o[dt][r] *= scr[r];

        // O += P @ V : V^T rows give kv chunks {0..7,16..23}/{8..15,24..31} per d-lane
#pragma unroll
        for (int dt = 0; dt < 4; ++dt) {
            Frag16 vf;
            const unsigned short* p = &Vh[(size_t)(dt * 16 + ln) * T_SEQ + kv0 + hh * 8];
            vf.q[0] = *(const uint4*)p;
            vf.q[1] = *(const uint4*)(p + 16);
            o[dt] = __builtin_amdgcn_wmma_f32_16x16x32_bf16(
                false, pf.v, false, vf.v, (short)0, o[dt], false, false);
        }
    }

    // finalize: divide by row sums, store bf16 to [B,T,H,D]
    float lr[8];
#pragma unroll
    for (int r = 0; r < 8; ++r) lr[r] = __shfl(lrow, hh * 8 + r, 32);
#pragma unroll
    for (int dt = 0; dt < 4; ++dt)
#pragma unroll
        for (int r = 0; r < 8; ++r) {
            int q = qbase + hh * 8 + r;
            int d = dt * 16 + ln;
            float val = o[dt][r] / lr[r];
            O[(((size_t)b * T_SEQ + q) * H_NUM + head) * 64 + d] = f2bf(val);
        }
}

// ---------------------------------------------------------------------------
// launcher
// ---------------------------------------------------------------------------
extern "C" void kernel_launch(void* const* d_in, const int* in_sizes, int n_in,
                              void* d_out, int out_size, void* d_ws, size_t ws_size,
                              hipStream_t stream) {
    const float* x    = (const float*)d_in[0];
    const float* wq   = (const float*)d_in[1];
    const float* wk   = (const float*)d_in[2];
    const float* wv   = (const float*)d_in[3];
    const float* wo   = (const float*)d_in[4];
    const float* cosT = (const float*)d_in[5];
    const float* sinT = (const float*)d_in[6];
    float* out = (float*)d_out;

    // workspace layout (bf16 element counts)
    const size_t X_E = (size_t)M_TOT * C_DIM;                 // 4,194,304
    const size_t W_E = (size_t)C_DIM * C_DIM;                 // 1,048,576
    const size_t Q_E = (size_t)B_SZ * H_NUM * T_SEQ * D_HEAD; // 4,194,304
    unsigned short* p   = (unsigned short*)d_ws;
    unsigned short* xb  = p;
    unsigned short* wqb = xb + X_E;
    unsigned short* wkb = wqb + W_E;
    unsigned short* wvb = wkb + W_E;
    unsigned short* wob = wvb + W_E;
    unsigned short* qb  = wob + W_E;
    unsigned short* kb  = qb + Q_E;
    unsigned short* vtb = kb + Q_E;
    unsigned short* ab  = vtb + Q_E;   // attention output [B,T,H,D] = [M,C]

    // 1) fp32 -> bf16 conversions
    f32_to_bf16_kernel<<<2048, 256, 0, stream>>>(x, xb, (int)X_E);
    f32_to_bf16_kernel<<<1024, 256, 0, stream>>>(wq, wqb, (int)W_E);
    f32_to_bf16_kernel<<<1024, 256, 0, stream>>>(wk, wkb, (int)W_E);
    f32_to_bf16_kernel<<<1024, 256, 0, stream>>>(wv, wvb, (int)W_E);
    f32_to_bf16_kernel<<<1024, 256, 0, stream>>>(wo, wob, (int)W_E);

    // 2) QKV projections (RoPE epilogue for Q/K, transposed store for V)
    dim3 ggrid(C_DIM / 128, M_TOT / 128);  // (8, 32)
    gemm_bf16_kernel<<<ggrid, 256, 0, stream>>>(xb, wqb, M_TOT, C_DIM, C_DIM,
                                                1, nullptr, qb, cosT, sinT);
    gemm_bf16_kernel<<<ggrid, 256, 0, stream>>>(xb, wkb, M_TOT, C_DIM, C_DIM,
                                                1, nullptr, kb, cosT, sinT);
    gemm_bf16_kernel<<<ggrid, 256, 0, stream>>>(xb, wvb, M_TOT, C_DIM, C_DIM,
                                                2, nullptr, vtb, cosT, sinT);

    // 3) causal flash attention
    dim3 fgrid(T_SEQ / 64, H_NUM, B_SZ);   // (32, 16, 2)
    flash_attn_kernel<<<fgrid, 128, 0, stream>>>(qb, kb, vtb, ab, 0.125f);

    // 4) output projection: out = attn @ wo.T  (f32 result)
    gemm_bf16_kernel<<<ggrid, 256, 0, stream>>>(ab, wob, M_TOT, C_DIM, C_DIM,
                                                0, out, nullptr, nullptr, nullptr);
}
